// MoELayer_68204080660481
// MI455X (gfx1250) — matrix operand order repaired and checked by reading, not compiled, not run on hardware
//
#include <hip/hip_runtime.h>
#include <hip/hip_bf16.h>
#include <math.h>

// Problem constants (from reference)
#define DIMX 2048
#define HIDX 2730
#define HID_P 2816            // HID padded to multiple of 64 (and 32 for WMMA K)
#define NEXP 8
#define NTOK 8192             // B*T
#define NK   (NTOK * 2)       // token-expert assignments (top-2)
#define CAP  (NK + NEXP * 128) // padded assignment capacity
#define LDA  40               // LDS row stride in halfs (80B: 16B-aligned, spreads banks)

typedef __bf16 bf16_t;
typedef __bf16 bf16x16 __attribute__((ext_vector_type(16)));
typedef __bf16 bf16x8  __attribute__((ext_vector_type(8)));
typedef float  floatx8 __attribute__((ext_vector_type(8)));

// ---------- WMMA fragment loaders (match CDNA5 ISA VGPR layouts) ----------
__device__ __forceinline__ bf16x16 frag_cat(bf16x8 lo, bf16x8 hi) {
    return __builtin_shufflevector(lo, hi, 0,1,2,3,4,5,6,7,8,9,10,11,12,13,14,15);
}
// A 16x32 bf16: lane r (0-15) h=0: K 0..7 & 16..23 ; lane r+16: K 8..15 & 24..31
__device__ __forceinline__ bf16x16 load_fragA(const bf16_t* tile, int lane) {
    const int r = lane & 15, h = lane >> 4;
    const bf16_t* q = tile + r * LDA + h * 8;
    return frag_cat(*(const bf16x8*)q, *(const bf16x8*)(q + 16));
}
// B 32x16 bf16 stored n-major in LDS: lane n (0-15): K 0..7 & 16..23 ; lane n+16: K 8..15 & 24..31
__device__ __forceinline__ bf16x16 load_fragB(const bf16_t* tile, int lane) {
    const int n = lane & 15, h = lane >> 4;
    const bf16_t* q = tile + n * LDA + h * 8;
    return frag_cat(*(const bf16x8*)q, *(const bf16x8*)(q + 16));
}
#define WMMA_BF16(A, B, C) \
    __builtin_amdgcn_wmma_f32_16x16x32_bf16(false, (A), false, (B), (short)0, (C), false, false)

// ---------- x fp32 -> bf16 (done once; reused across all GEMM K-loops) ----------
__global__ __launch_bounds__(256) void cvt_kernel(const float* __restrict__ x,
                                                  bf16_t* __restrict__ xb) {
    const size_t i = ((size_t)blockIdx.x * 256 + threadIdx.x) * 4;
    const float4 v = *(const float4*)(x + i);
    bf16_t t[4] = { (bf16_t)v.x, (bf16_t)v.y, (bf16_t)v.z, (bf16_t)v.w };
    *(uint2*)(xb + i) = *(const uint2*)t;
}

// ---------- router: logits -> top2 -> renormalized softmax weights ----------
__global__ __launch_bounds__(256) void router_kernel(const float* __restrict__ x,
                                                     const float* __restrict__ rw,
                                                     int* __restrict__ topi,
                                                     float* __restrict__ topv) {
    const int lane = threadIdx.x & 31;
    const int t = blockIdx.x * 8 + (threadIdx.x >> 5);
    float acc[NEXP];
#pragma unroll
    for (int e = 0; e < NEXP; ++e) acc[e] = 0.f;
    const float* xr = x + (size_t)t * DIMX;
    for (int d = lane; d < DIMX; d += 32) {
        const float xv = xr[d];
#pragma unroll
        for (int e = 0; e < NEXP; ++e) acc[e] += xv * rw[d * NEXP + e];
    }
#pragma unroll
    for (int e = 0; e < NEXP; ++e)
        for (int off = 16; off > 0; off >>= 1) acc[e] += __shfl_xor(acc[e], off, 32);
    if (lane == 0) {
        int i0 = 0; float l0 = acc[0];
        for (int e = 1; e < NEXP; ++e) if (acc[e] > l0) { l0 = acc[e]; i0 = e; }
        int i1 = (i0 == 0) ? 1 : 0; float l1 = acc[i1];
        for (int e = 0; e < NEXP; ++e) {
            if (e == i0) continue;
            if (acc[e] > l1) { l1 = acc[e]; i1 = e; }
        }
        const float v0 = 1.f / (1.f + __expf(l1 - l0));  // softmax over top2
        topi[t * 2] = i0; topi[t * 2 + 1] = i1;
        topv[t * 2] = v0; topv[t * 2 + 1] = 1.f - v0;
    }
}

// ---------- per-expert counts and 128-padded segment offsets ----------
// ctrl[0..7]=cnt, ctrl[8..15]=cntPad, ctrl[16..23]=poff, ctrl[24]=totalPad
__global__ __launch_bounds__(256) void count_kernel(const int* __restrict__ topi,
                                                    int* __restrict__ ctrl,
                                                    int* __restrict__ idxArr,
                                                    float* __restrict__ wgtArr) {
    __shared__ int sc[NEXP];
    const int tid = threadIdx.x;
    if (tid < NEXP) sc[tid] = 0;
    __syncthreads();
    for (int i = tid; i < NK; i += 256) atomicAdd(&sc[topi[i]], 1);
    for (int i = tid; i < CAP; i += 256) { idxArr[i] = 0; wgtArr[i] = 0.f; }
    __syncthreads();
    if (tid == 0) {
        int off = 0;
        for (int e = 0; e < NEXP; ++e) {
            const int c = sc[e];
            const int cp = (c + 127) & ~127;
            ctrl[e] = c; ctrl[8 + e] = cp; ctrl[16 + e] = off;
            off += cp;
        }
        ctrl[24] = off;
    }
}

// ---------- deterministic gather-list build: wave e scans all assignments ----------
__global__ __launch_bounds__(256) void assign_kernel(const int* __restrict__ topi,
                                                     const float* __restrict__ topv,
                                                     const int* __restrict__ ctrl,
                                                     int* __restrict__ idxArr,
                                                     float* __restrict__ wgtArr,
                                                     int* __restrict__ slotOf) {
    const int lane = threadIdx.x & 31;
    const int e = threadIdx.x >> 5;  // one wave per expert
    const int base0 = ctrl[16 + e];
    const unsigned ltmask = (1u << lane) - 1u;
    int running = 0;
    for (int b = 0; b < NK; b += 32) {
        const int i = b + lane;
        const bool match = (topi[i] == e);
        const unsigned mask = (unsigned)__ballot(match);
        if (match) {
            const int s = base0 + running + __popc(mask & ltmask);
            idxArr[s] = i >> 1;       // token
            wgtArr[s] = topv[i];
            slotOf[i] = s;
        }
        running += __popc(mask);
    }
}

// ---------- GEMM1: h = silu(x*w1) * (x*w3), bf16 out, block 128x64, wave 32x32 ----------
template <bool EXPERT>
__global__ __launch_bounds__(256) void gemm1_kernel(const bf16_t* __restrict__ xb,
                                                    const float* __restrict__ w1g,
                                                    const float* __restrict__ w3g,
                                                    bf16_t* __restrict__ h,
                                                    const int* __restrict__ ctrl,
                                                    const int* __restrict__ idxArr) {
    const int nt = blockIdx.x;   // HID tile (64 wide)
    const int mt = blockIdx.y;   // row tile (128)
    int rowBase = 0;
    const float* W1 = w1g;
    const float* W3 = w3g;
    if (EXPERT) {
        const int e = blockIdx.z;
        if (mt * 128 >= ctrl[8 + e]) return;  // beyond padded count: idle block
        rowBase = ctrl[16 + e];
        W1 = w1g + (size_t)e * DIMX * HIDX;
        W3 = w3g + (size_t)e * DIMX * HIDX;
    }
    __shared__ __attribute__((aligned(16))) bf16_t As[128 * LDA];
    __shared__ __attribute__((aligned(16))) bf16_t B1s[64 * LDA];
    __shared__ __attribute__((aligned(16))) bf16_t B3s[64 * LDA];

    const int tid = threadIdx.x, lane = tid & 31, wv = tid >> 5;
    const int wm = wv >> 1, wn = wv & 1;  // 4x2 waves, wave tile 32(m) x 32(n)

    floatx8 acc1[2][2], acc3[2][2];
#pragma unroll
    for (int a = 0; a < 2; ++a)
#pragma unroll
        for (int b = 0; b < 2; ++b) {
            acc1[a][b] = (floatx8){0,0,0,0,0,0,0,0};
            acc3[a][b] = (floatx8){0,0,0,0,0,0,0,0};
        }

    const int arow = tid >> 1, apart = tid & 1;
    const int atok = EXPERT ? idxArr[rowBase + mt * 128 + arow] : (mt * 128 + arow);
    const bf16_t* asrc = xb + (size_t)atok * DIMX + apart * 16;
    bf16_t* adst = &As[arow * LDA + apart * 16];

    for (int k0 = 0; k0 < DIMX; k0 += 32) {
        // stage A (already bf16): 32B per thread
        *(uint4*)adst       = *(const uint4*)(asrc + k0);
        *(uint4*)(adst + 8) = *(const uint4*)(asrc + k0 + 8);
        // stage B1/B3: fp32 -> bf16, transpose to n-major, guard n < HID
#pragma unroll
        for (int i = 0; i < 8; ++i) {
            const int li = tid + i * 256;
            const int n = li & 63, k = li >> 6;
            const int gn = nt * 64 + n;
            float v1 = 0.f, v3 = 0.f;
            if (gn < HIDX) {
                const size_t g = (size_t)(k0 + k) * HIDX + gn;
                v1 = W1[g]; v3 = W3[g];
            }
            B1s[n * LDA + k] = (bf16_t)v1;
            B3s[n * LDA + k] = (bf16_t)v3;
        }
        if (k0 + 32 < DIMX) {
            __builtin_prefetch(W1 + (size_t)(k0 + 32) * HIDX + nt * 64, 0, 1);
            __builtin_prefetch(W3 + (size_t)(k0 + 32) * HIDX + nt * 64, 0, 1);
        }
        __syncthreads();
        const bf16x16 a0 = load_fragA(&As[(wm * 32) * LDA], lane);
        const bf16x16 a1 = load_fragA(&As[(wm * 32 + 16) * LDA], lane);
        const bf16x16 b10 = load_fragB(&B1s[(wn * 32) * LDA], lane);
        const bf16x16 b11 = load_fragB(&B1s[(wn * 32 + 16) * LDA], lane);
        const bf16x16 b30 = load_fragB(&B3s[(wn * 32) * LDA], lane);
        const bf16x16 b31 = load_fragB(&B3s[(wn * 32 + 16) * LDA], lane);
        acc1[0][0] = WMMA_BF16(a0, b10, acc1[0][0]);
        acc1[0][1] = WMMA_BF16(a0, b11, acc1[0][1]);
        acc1[1][0] = WMMA_BF16(a1, b10, acc1[1][0]);
        acc1[1][1] = WMMA_BF16(a1, b11, acc1[1][1]);
        acc3[0][0] = WMMA_BF16(a0, b30, acc3[0][0]);
        acc3[0][1] = WMMA_BF16(a0, b31, acc3[0][1]);
        acc3[1][0] = WMMA_BF16(a1, b30, acc3[1][0]);
        acc3[1][1] = WMMA_BF16(a1, b31, acc3[1][1]);
        __syncthreads();
    }
    // epilogue: silu(c1)*c3 -> bf16 h (HID_P stride; padded cols are exactly 0)
#pragma unroll
    for (int ms = 0; ms < 2; ++ms)
#pragma unroll
        for (int ns = 0; ns < 2; ++ns)
#pragma unroll
            for (int j = 0; j < 8; ++j) {
                const int m = ((lane >> 4) << 3) + j;
                const int n = lane & 15;
                const float c1 = acc1[ms][ns][j], c3 = acc3[ms][ns][j];
                const float v = (c1 / (1.f + __expf(-c1))) * c3;
                const size_t row = (size_t)(rowBase + mt * 128 + wm * 32 + ms * 16 + m);
                const int col = nt * 64 + wn * 32 + ns * 16 + n;
                h[row * HID_P + col] = (bf16_t)v;
            }
}

// ---------- GEMM2: out = h * w2 (+ weight-scale epilogue), block 128x128, wave 32x64 ----------
template <bool EXPERT>
__global__ __launch_bounds__(256) void gemm2_kernel(const bf16_t* __restrict__ h,
                                                    const float* __restrict__ w2g,
                                                    float* __restrict__ out,
                                                    bf16_t* __restrict__ outg,
                                                    const float* __restrict__ wgtArr,
                                                    const int* __restrict__ ctrl) {
    const int nt = blockIdx.x;  // DIM tile (128)
    const int mt = blockIdx.y;  // row tile (128)
    int rowBase = 0;
    const float* W2 = w2g;
    if (EXPERT) {
        const int e = blockIdx.z;
        if (mt * 128 >= ctrl[8 + e]) return;
        rowBase = ctrl[16 + e];
        W2 = w2g + (size_t)e * HIDX * DIMX;
    }
    __shared__ __attribute__((aligned(16))) bf16_t As[128 * LDA];
    __shared__ __attribute__((aligned(16))) bf16_t Bs[128 * LDA];

    const int tid = threadIdx.x, lane = tid & 31, wv = tid >> 5;
    const int wm = wv >> 1, wn = wv & 1;  // wave tile 32(m) x 64(n)

    floatx8 acc[2][4];
#pragma unroll
    for (int a = 0; a < 2; ++a)
#pragma unroll
        for (int b = 0; b < 4; ++b) acc[a][b] = (floatx8){0,0,0,0,0,0,0,0};

    const int arow = tid >> 1, apart = tid & 1;
    const bf16_t* asrc = h + (size_t)(rowBase + mt * 128 + arow) * HID_P + apart * 16;
    bf16_t* adst = &As[arow * LDA + apart * 16];

    for (int k0 = 0; k0 < HID_P; k0 += 32) {
        *(uint4*)adst       = *(const uint4*)(asrc + k0);
        *(uint4*)(adst + 8) = *(const uint4*)(asrc + k0 + 8);
#pragma unroll
        for (int i = 0; i < 16; ++i) {
            const int li = tid + i * 256;
            const int n = li & 127, k = li >> 7;
            const int gk = k0 + k;
            const float v = (gk < HIDX) ? W2[(size_t)gk * DIMX + nt * 128 + n] : 0.f;
            Bs[n * LDA + k] = (bf16_t)v;
        }
        if (k0 + 32 < HIDX)
            __builtin_prefetch(W2 + (size_t)(k0 + 32) * DIMX + nt * 128, 0, 1);
        __syncthreads();
        const bf16x16 a0 = load_fragA(&As[(wm * 32) * LDA], lane);
        const bf16x16 a1 = load_fragA(&As[(wm * 32 + 16) * LDA], lane);
#pragma unroll
        for (int ns = 0; ns < 4; ++ns) {
            const bf16x16 b = load_fragB(&Bs[(wn * 64 + ns * 16) * LDA], lane);
            acc[0][ns] = WMMA_BF16(a0, b, acc[0][ns]);
            acc[1][ns] = WMMA_BF16(a1, b, acc[1][ns]);
        }
        __syncthreads();
    }
#pragma unroll
    for (int ms = 0; ms < 2; ++ms)
#pragma unroll
        for (int ns = 0; ns < 4; ++ns)
#pragma unroll
            for (int j = 0; j < 8; ++j) {
                const int m = ((lane >> 4) << 3) + j;
                const int n = lane & 15;
                const size_t row = (size_t)(rowBase + mt * 128 + wm * 32 + ms * 16 + m);
                const size_t col = (size_t)(nt * 128 + wn * 64 + ns * 16 + n);
                if (EXPERT) {
                    outg[row * DIMX + col] = (bf16_t)(wgtArr[row] * acc[ms][ns][j]);
                } else {
                    out[row * DIMX + col] = acc[ms][ns][j];
                }
            }
}

// ---------- atomic-free combine: out = shared + outg[slot0] + outg[slot1] ----------
__global__ __launch_bounds__(256) void combine_kernel(float* __restrict__ out,
                                                      const bf16_t* __restrict__ outg,
                                                      const int* __restrict__ slotOf) {
    const int t = blockIdx.x;
    const size_t s0 = (size_t)slotOf[t * 2];
    const size_t s1 = (size_t)slotOf[t * 2 + 1];
    float* orow = out + (size_t)t * DIMX;
    const bf16_t* g0 = outg + s0 * DIMX;
    const bf16_t* g1 = outg + s1 * DIMX;
    for (int i = threadIdx.x; i < DIMX; i += 256)
        orow[i] += (float)g0[i] + (float)g1[i];
}

extern "C" void kernel_launch(void* const* d_in, const int* in_sizes, int n_in,
                              void* d_out, int out_size, void* d_ws, size_t ws_size,
                              hipStream_t stream) {
    const float* x   = (const float*)d_in[0];
    const float* rw  = (const float*)d_in[1];
    const float* sw1 = (const float*)d_in[2];
    const float* sw2 = (const float*)d_in[3];
    const float* sw3 = (const float*)d_in[4];
    const float* ew1 = (const float*)d_in[5];
    const float* ew2 = (const float*)d_in[6];
    const float* ew3 = (const float*)d_in[7];
    float* out = (float*)d_out;

    char* ws = (char*)d_ws;
    size_t off = 0;
    auto alloc = [&](size_t bytes) -> void* {
        void* p = ws + off;
        off = (off + bytes + 255) & ~(size_t)255;
        return p;
    };
    int*    topi   = (int*)   alloc((size_t)NK * 4);
    float*  topv   = (float*) alloc((size_t)NK * 4);
    int*    slotOf = (int*)   alloc((size_t)NK * 4);
    int*    ctrl   = (int*)   alloc(256);
    int*    idxArr = (int*)   alloc((size_t)CAP * 4);
    float*  wgtArr = (float*) alloc((size_t)CAP * 4);
    bf16_t* xb     = (bf16_t*)alloc((size_t)NTOK * DIMX * 2);   // 32 MB
    bf16_t* hbuf   = (bf16_t*)alloc((size_t)CAP * HID_P * 2);   // ~94 MB
    bf16_t* outg   = (bf16_t*)alloc((size_t)CAP * DIMX * 2);    // ~68 MB
    (void)in_sizes; (void)n_in; (void)out_size; (void)ws_size;

    cvt_kernel<<<(NTOK * DIMX) / 1024, 256, 0, stream>>>(x, xb);
    router_kernel<<<NTOK / 8, 256, 0, stream>>>(x, rw, topi, topv);
    count_kernel<<<1, 256, 0, stream>>>(topi, ctrl, idxArr, wgtArr);
    assign_kernel<<<1, 256, 0, stream>>>(topi, topv, ctrl, idxArr, wgtArr, slotOf);

    // shared expert: writes d_out directly
    gemm1_kernel<false><<<dim3(HID_P / 64, NTOK / 128, 1), 256, 0, stream>>>(
        xb, sw1, sw3, hbuf, ctrl, idxArr);
    gemm2_kernel<false><<<dim3(DIMX / 128, NTOK / 128, 1), 256, 0, stream>>>(
        hbuf, sw2, out, outg, wgtArr, ctrl);

    // routed experts over gathered rows (blocks past padded count exit early)
    gemm1_kernel<true><<<dim3(HID_P / 64, NTOK / 128, NEXP), 256, 0, stream>>>(
        xb, ew1, ew3, hbuf, ctrl, idxArr);
    gemm2_kernel<true><<<dim3(DIMX / 128, NTOK / 128, NEXP), 256, 0, stream>>>(
        hbuf, ew2, out, outg, wgtArr, ctrl);

    combine_kernel<<<NTOK, 256, 0, stream>>>(out, outg, slotOf);
}
